// single_generator_block_45981919871715
// MI455X (gfx1250) — matrix operand order repaired and checked by reading, not compiled, not run on hardware
//
#include <hip/hip_runtime.h>
#include <hip/hip_bf16.h>
#include <stdint.h>

typedef __attribute__((ext_vector_type(16))) __bf16 v16bf;
typedef __attribute__((ext_vector_type(8)))  float  v8f;

#define B_TOT   8192
#define NZ_     100
#define H1      256
#define H2      64
#define FNUM    8192
#define FDIM    1024
#define BN_EPS  1e-5f
#define GEPS    0.001f
#define INV_T   10.0f

// workspace layout (float offsets)
#define OFF_SUM1 0
#define OFF_SSQ1 256
#define OFF_S1   512
#define OFF_T1   768
#define OFF_SUM2 1024
#define OFF_SSQ2 1088
#define OFF_S2   1152
#define OFF_T2   1216
#define OFF_A1   1280
#define OFF_A2   (OFF_A1 + B_TOT*H1)          // 8192*256 f32
#define OFF_W3T  (OFF_A2 + B_TOT*H2)          // after: 8192*64 bf16 region

// ---------------- GEMM1: z = [x|y] @ W1 + b1, accumulate batch stats ----------------
__global__ void k_gemm1(const float* __restrict__ x, const float* __restrict__ y,
                        const float* __restrict__ W1, const float* __restrict__ b1,
                        float* __restrict__ ws) {
  const int j  = threadIdx.x;             // output column 0..255
  const int r0 = blockIdx.x * 32;
  float ls = 0.f, lss = 0.f;
  for (int r = r0; r < r0 + 32; ++r) {
    const float* xr = x + (size_t)r * NZ_;
    float acc = b1[j];
    #pragma unroll 4
    for (int k = 0; k < NZ_; ++k) acc = fmaf(xr[k], W1[k*H1 + j], acc);
    acc = fmaf(y[r*2+0], W1[100*H1 + j], acc);
    acc = fmaf(y[r*2+1], W1[101*H1 + j], acc);
    ws[OFF_A1 + (size_t)r*H1 + j] = acc;
    ls += acc; lss += acc*acc;
  }
  atomicAdd(&ws[OFF_SUM1 + j], ls);
  atomicAdd(&ws[OFF_SSQ1 + j], lss);
}

// ---------------- fold BN into per-column scale/shift ----------------
__global__ void k_bnparams(float* __restrict__ ws, const float* __restrict__ g,
                           const float* __restrict__ be, int n,
                           int offSum, int offSsq, int offS, int offT) {
  int j = threadIdx.x;
  if (j < n) {
    float mu  = ws[offSum + j] * (1.0f / B_TOT);
    float var = ws[offSsq + j] * (1.0f / B_TOT) - mu * mu;
    float s   = g[j] * rsqrtf(var + BN_EPS);
    ws[offS + j] = s;
    ws[offT + j] = be[j] - mu * s;
  }
}

// ---------------- GEMM2: relu(bn(a1)) @ W2 + b2, accumulate batch stats ----------------
__global__ void k_gemm2(const float* __restrict__ W2, const float* __restrict__ b2,
                        float* __restrict__ ws) {
  __shared__ float s1[H1], t1[H1];
  const int t = threadIdx.x;
  s1[t] = ws[OFF_S1 + t];
  t1[t] = ws[OFF_T1 + t];
  __syncthreads();
  const int j  = t & 63;
  const int rs = t >> 6;                  // 0..3
  const int r0 = blockIdx.x * 64;
  float ls = 0.f, lss = 0.f;
  for (int ri = 0; ri < 16; ++ri) {
    const int r = r0 + rs + ri * 4;
    const float* a1r = ws + OFF_A1 + (size_t)r * H1;
    float acc = b2[j];
    #pragma unroll 4
    for (int k = 0; k < H1; ++k) {
      float h = fmaxf(fmaf(a1r[k], s1[k], t1[k]), 0.f);
      acc = fmaf(h, W2[k*H2 + j], acc);
    }
    ws[OFF_A2 + (size_t)r*H2 + j] = acc;
    ls += acc; lss += acc*acc;
  }
  atomicAdd(&ws[OFF_SUM2 + j], ls);
  atomicAdd(&ws[OFF_SSQ2 + j], lss);
}

// ---------------- W3 [64,8192] f32 -> W3T [8192,64] bf16 ----------------
__global__ void k_w3cvt(const float* __restrict__ W3, float* __restrict__ ws) {
  __bf16* w3t = (__bf16*)(ws + OFF_W3T);
  int idx = blockIdx.x * 256 + threadIdx.x;     // idx = n*64 + k
  int n = idx >> 6, k = idx & 63;
  w3t[idx] = (__bf16)W3[(size_t)k * FNUM + n];
}

// ---------------- cheap 32-bit hash -> uniform -> gumbel ----------------
__device__ __forceinline__ float gumbel_noise(uint32_t idx) {
  uint32_t h = idx + 0x9E377942u;               // "key 42" flavored offset
  h ^= h >> 16; h *= 0x21F0AAADu;
  h ^= h >> 15; h *= 0xD35A2D97u;
  h ^= h >> 15;
  float u = (float)h * 2.3283064e-10f;          // [0,1)
  float inner = GEPS - __logf(fmaf(u, GEPS, GEPS));   // -log(eps*U+eps)+eps
  return -__logf(inner);
}

// ---------------- fused GEMM3 + gumbel + online softmax/argmax + codebook gather ----------------
__launch_bounds__(256)
__global__ void k_logits_argmax_gather(const float* __restrict__ b3,
                                       const float* __restrict__ codebook,
                                       const float* __restrict__ ws,
                                       float* __restrict__ out) {
  __shared__ __bf16 sA[16 * 64];
  __shared__ float pMax[8][16], pSum[8][16];
  __shared__ int   pArg[8][16];
  __shared__ int   sInd[16];
  __shared__ float sVal[16];

  const __bf16* w3t = (const __bf16*)(ws + OFF_W3T);
  const int t    = threadIdx.x;
  const int wave = t >> 5;
  const int lane = t & 31;
  const int b0   = blockIdx.x * 16;      // 16 batch rows per block

  // stage A tile: bf16(relu(bn(a2))) [16 x 64]
  for (int i = t; i < 16 * 64; i += 256) {
    int r = i >> 6, c = i & 63;
    float av = ws[OFF_A2 + (size_t)(b0 + r) * H2 + c];
    float h  = fmaxf(fmaf(av, ws[OFF_S2 + c], ws[OFF_T2 + c]), 0.f);
    sA[i] = (__bf16)h;
  }
  __syncthreads();

  // A fragments, contiguous K order matching B: lane m=lane&15; K = e + 16*(lane>>4)
  const int m   = lane & 15;
  const int khi = (lane >> 4) * 16;
  v16bf a0, a1;
  #pragma unroll
  for (int e = 0; e < 16; ++e) {
    int k = e + khi;
    a0[e] = sA[m * 64 + k];
    a1[e] = sA[m * 64 + 32 + k];
  }

  // online softmax state: VGPR slot v holds row (v + 8*(lane>=16)), column N = lane&15 per tile
  float rmax[8], rsum[8]; int rarg[8];
  #pragma unroll
  for (int v = 0; v < 8; ++v) { rmax[v] = -__builtin_inff(); rsum[v] = 0.f; rarg[v] = 0; }

  const int nlow   = lane & 15;
  const int kb     = (lane >> 4) * 16;   // B layout: lanes 0-15 K=0..15, lanes 16-31 K=16..31
  const int rowOff = (lane >> 4) * 8;
  const int NT     = FNUM / 16;          // 512 N-tiles; each wave runs 64 -> unroll x2, no tail

  for (int tile = wave; tile < NT; tile += 16) {
    const int tileB = tile + 8;          // always < NT (stride-16 loop, 64 tiles/wave, even)
    const __bf16* wpA = w3t + (size_t)(tile  * 16 + nlow) * 64;
    const __bf16* wpB = w3t + (size_t)(tileB * 16 + nlow) * 64;
    v16bf bA0 = *(const v16bf*)(wpA + kb);
    v16bf bA1 = *(const v16bf*)(wpA + 32 + kb);
    v16bf bB0 = *(const v16bf*)(wpB + kb);
    v16bf bB1 = *(const v16bf*)(wpB + 32 + kb);
    float biasA = b3[tile * 16 + nlow];
    float biasB = b3[tileB * 16 + nlow];
    __builtin_prefetch(wpB + 8 * 16 * 64, 0, 0);       // tile+16 -> global_prefetch_b8

    // ---- sub-tile A ----
    {
      v8f c = {};
      c = __builtin_amdgcn_wmma_f32_16x16x32_bf16(false, a0, false, bA0, (short)0, c, false, false);
      c = __builtin_amdgcn_wmma_f32_16x16x32_bf16(false, a1, false, bA1, (short)0, c, false, false);
      const int f = tile * 16 + nlow;
      #pragma unroll
      for (int v = 0; v < 8; ++v) {
        const uint32_t idx = (uint32_t)((b0 + v + rowOff) << 13) + (uint32_t)f;  // b*8192+f
        float lv = (c[v] + biasA + gumbel_noise(idx)) * INV_T;
        float d  = lv - rmax[v];
        float e  = __expf(-__builtin_fabsf(d));        // exp(d) if d<=0 else exp(-d)
        bool  gt = d > 0.f;
        rsum[v] = gt ? fmaf(rsum[v], e, 1.f) : (rsum[v] + e);
        rarg[v] = gt ? f : rarg[v];
        rmax[v] = gt ? lv : rmax[v];
      }
    }
    // ---- sub-tile B ----
    {
      v8f c = {};
      c = __builtin_amdgcn_wmma_f32_16x16x32_bf16(false, a0, false, bB0, (short)0, c, false, false);
      c = __builtin_amdgcn_wmma_f32_16x16x32_bf16(false, a1, false, bB1, (short)0, c, false, false);
      const int f = tileB * 16 + nlow;
      #pragma unroll
      for (int v = 0; v < 8; ++v) {
        const uint32_t idx = (uint32_t)((b0 + v + rowOff) << 13) + (uint32_t)f;
        float lv = (c[v] + biasB + gumbel_noise(idx)) * INV_T;
        float d  = lv - rmax[v];
        float e  = __expf(-__builtin_fabsf(d));
        bool  gt = d > 0.f;
        rsum[v] = gt ? fmaf(rsum[v], e, 1.f) : (rsum[v] + e);
        rarg[v] = gt ? f : rarg[v];
        rmax[v] = gt ? lv : rmax[v];
      }
    }
  }

  // reduce across the 16 lanes that share a row (xor masks stay inside each 16-lane half)
  #pragma unroll
  for (int v = 0; v < 8; ++v) {
    #pragma unroll
    for (int off = 1; off < 16; off <<= 1) {
      float om = __shfl_xor(rmax[v], off, 32);
      float os = __shfl_xor(rsum[v], off, 32);
      int   oa = __shfl_xor(rarg[v], off, 32);
      float d  = om - rmax[v];
      float e  = __expf(-__builtin_fabsf(d));
      bool  gt = d > 0.f;
      bool  tk = gt || (d == 0.f && oa < rarg[v]);
      rsum[v] = gt ? fmaf(rsum[v], e, os) : fmaf(os, e, rsum[v]);
      rarg[v] = tk ? oa : rarg[v];
      rmax[v] = fmaxf(rmax[v], om);
    }
  }
  if ((lane & 15) == 0) {
    #pragma unroll
    for (int v = 0; v < 8; ++v) {
      pMax[wave][rowOff + v] = rmax[v];
      pSum[wave][rowOff + v] = rsum[v];
      pArg[wave][rowOff + v] = rarg[v];
    }
  }
  __syncthreads();

  // merge the 8 wave partials per row
  if (t < 16) {
    float M = -__builtin_inff();
    #pragma unroll
    for (int w = 0; w < 8; ++w) M = fmaxf(M, pMax[w][t]);
    float S = 0.f; int A = 0x7FFFFFFF;
    #pragma unroll
    for (int w = 0; w < 8; ++w) {
      S += pSum[w][t] * __expf(pMax[w][t] - M);
      if (pMax[w][t] == M && pArg[w][t] < A) A = pArg[w][t];
    }
    float sprob = 1.0f / S;                    // softmax prob of the max element
    sInd[t] = A;
    sVal[t] = (1.0f - sprob) + sprob;          // straight-through forward value at argmax
    __builtin_prefetch(codebook + (size_t)A * FDIM, 0, 0);
  }
  __syncthreads();

  // out[row,:] = val * codebook[ind,:]  (256 threads x float4 == 1024 floats)
  const float4* cb4  = (const float4*)codebook;
  float4*       out4 = (float4*)out;
  for (int r = 0; r < 16; ++r) {
    const int   ind = sInd[r];
    const float val = sVal[r];
    float4 cv = cb4[(size_t)ind * (FDIM / 4) + t];
    float4 ov; ov.x = val * cv.x; ov.y = val * cv.y; ov.z = val * cv.z; ov.w = val * cv.w;
    out4[(size_t)(b0 + r) * (FDIM / 4) + t] = ov;
  }
}

extern "C" void kernel_launch(void* const* d_in, const int* in_sizes, int n_in,
                              void* d_out, int out_size, void* d_ws, size_t ws_size,
                              hipStream_t stream) {
  const float* x   = (const float*)d_in[0];
  const float* y   = (const float*)d_in[1];
  const float* W1  = (const float*)d_in[2];
  const float* b1  = (const float*)d_in[3];
  const float* g1  = (const float*)d_in[4];
  const float* be1 = (const float*)d_in[5];
  const float* W2  = (const float*)d_in[6];
  const float* b2  = (const float*)d_in[7];
  const float* g2  = (const float*)d_in[8];
  const float* be2 = (const float*)d_in[9];
  const float* W3  = (const float*)d_in[10];
  const float* b3  = (const float*)d_in[11];
  const float* cb  = (const float*)d_in[12];
  float* ws  = (float*)d_ws;
  float* out = (float*)d_out;
  (void)in_sizes; (void)n_in; (void)out_size; (void)ws_size;

  hipMemsetAsync(d_ws, 0, 1280 * sizeof(float), stream);  // zero BN accumulators
  k_gemm1<<<B_TOT / 32, 256, 0, stream>>>(x, y, W1, b1, ws);
  k_bnparams<<<1, 256, 0, stream>>>(ws, g1, be1, H1, OFF_SUM1, OFF_SSQ1, OFF_S1, OFF_T1);
  k_gemm2<<<B_TOT / 64, 256, 0, stream>>>(W2, b2, ws);
  k_bnparams<<<1, 256, 0, stream>>>(ws, g2, be2, H2, OFF_SUM2, OFF_SSQ2, OFF_S2, OFF_T2);
  k_w3cvt<<<(FNUM * H2) / 256, 256, 0, stream>>>(W3, ws);
  k_logits_argmax_gather<<<B_TOT / 16, 256, 0, stream>>>(b3, cb, ws, out);
}
